// RnnSearchModule_76673756168562
// MI455X (gfx1250) — matrix operand order repaired and checked by reading, not compiled, not run on hardware
//
#include <hip/hip_runtime.h>
#include <hip/hip_bf16.h>

// ---------------------------------------------------------------------------
// RNN search cell: s = tanh(x@Wx + h@Wh + b); BatchNorm over batch; norm clip.
// T=256, B=128, C=512. fp32 in/out; bf16 WMMA (f32 accum) for both GEMMs.
// Phase 1 stages the WMMA B-operand into LDS with global_load_async_to_lds.
// ---------------------------------------------------------------------------

typedef __attribute__((ext_vector_type(16))) __bf16 v16bf;
typedef __attribute__((ext_vector_type(8)))  __bf16 v8bf;
typedef __attribute__((ext_vector_type(8)))  float  v8f;

#define T_   256
#define B_   128
#define C_   512
#define TBC_ (T_*B_*C_)
#define HPAD 520   // padded LDS row stride in bf16 elems (1040B, 16B aligned, 4-bank skew)

union U16B { uint4 u; v8bf h; };

static __device__ __forceinline__ v8bf load8bf(const __bf16* p) {
    U16B c; c.u = *(const uint4*)p; return c.h;
}
static __device__ __forceinline__ v16bf cat16(v8bf lo, v8bf hi) {
    return __builtin_shufflevector(lo, hi, 0,1,2,3,4,5,6,7,8,9,10,11,12,13,14,15);
}
static __device__ __forceinline__ float fast_tanh(float x) {
    // tanh(x) = 1 - 2/(exp(2x)+1); safe at +/-inf of exp
    float e = __expf(2.0f * x);
    return 1.0f - 2.0f / (e + 1.0f);
}
// Issue one per-lane async 16B global->LDS copy (ASYNCcnt-tracked).
static __device__ __forceinline__ void async_g2l_b128(const void* gptr, void* lptr) {
    unsigned lds_addr = (unsigned)(uintptr_t)lptr;                 // low 32b = LDS offset
    unsigned long long gaddr = (unsigned long long)(uintptr_t)gptr;
    asm volatile("global_load_async_to_lds_b128 %0, %1, off"
                 :: "v"(lds_addr), "v"(gaddr) : "memory");
}
static __device__ __forceinline__ void wait_async0() {
    asm volatile("s_wait_asynccnt 0x0" ::: "memory");
}

// ---------------------------------------------------------------------------
// Kernel 0: transpose + convert weights to bf16:  WT[n*C+k] = bf16(W[k*C+n])
// ---------------------------------------------------------------------------
__global__ void convert_weights(const float* __restrict__ Wx,
                                const float* __restrict__ Wh,
                                __bf16* __restrict__ WxT,
                                __bf16* __restrict__ WhT) {
    int i = blockIdx.x * blockDim.x + threadIdx.x;   // 0 .. C*C-1
    int k = i >> 9, n = i & (C_ - 1);
    WxT[n * C_ + k] = (__bf16)Wx[k * C_ + n];
    WhT[n * C_ + k] = (__bf16)Wh[k * C_ + n];
}

// ---------------------------------------------------------------------------
// Kernel 1: P[TB, C] = X[TB, C] @ Wx + b     (parallel over all T -- no dep)
// Block = 8 waves sharing one 16-column B slab (async-staged to LDS once);
// each wave owns one 16-row M tile. K=512 in steps of 32 via
// v_wmma_f32_16x16x32_bf16. A converted fp32->bf16 in-register.
// ---------------------------------------------------------------------------
__global__ __launch_bounds__(256) void gemm_xwx(
    const float* __restrict__ X,      // [T*B, C] fp32 row-major
    const __bf16* __restrict__ WxT,   // [C(n), C(k)] bf16
    const float* __restrict__ bvec,   // [C]
    float* __restrict__ P)            // [T*B, C] fp32
{
    __shared__ __bf16 bslab[16 * HPAD];            // 16 cols x 512 K, padded

    const int lane  = threadIdx.x & 31;
    const int wave  = threadIdx.x >> 5;
    const int nt    = blockIdx.x & 31;             // column tile (shared by block)
    const int mt    = (blockIdx.x >> 5) * 8 + wave;// row tile (per wave)
    const int row   = mt * 16 + (lane & 15);
    const int col   = nt * 16 + (lane & 15);
    const int khalf = lane >> 4;

    // ---- async-stage B slab: WxT rows [nt*16, nt*16+16) -> LDS (16KB)
    {
        const __bf16* gbase = WxT + nt * 16 * C_;
        for (int i = threadIdx.x; i < 1024; i += 256) {   // 1024 x 16B chunks
            int j = i >> 6;                                // slab row 0..15
            int o = (i & 63) * 8;                          // bf16 offset in row
            async_g2l_b128(gbase + j * C_ + o, &bslab[j * HPAD + o]);
        }
        wait_async0();
        __syncthreads();
    }

    v8f acc = {};
    for (int kb = 0; kb < C_; kb += 32) {
        // ---- A fragment (16x32 bf16): elems 0..7 @ k0, 8..15 @ k0+16
        const float* ap = X + row * C_ + kb + khalf * 8;
        float4 f0 = *(const float4*)(ap + 0);
        float4 f1 = *(const float4*)(ap + 4);
        float4 f2 = *(const float4*)(ap + 16);
        float4 f3 = *(const float4*)(ap + 20);
        v16bf a;
        a[0]=(__bf16)f0.x; a[1]=(__bf16)f0.y; a[2] =(__bf16)f0.z; a[3] =(__bf16)f0.w;
        a[4]=(__bf16)f1.x; a[5]=(__bf16)f1.y; a[6] =(__bf16)f1.z; a[7] =(__bf16)f1.w;
        a[8]=(__bf16)f2.x; a[9]=(__bf16)f2.y; a[10]=(__bf16)f2.z; a[11]=(__bf16)f2.w;
        a[12]=(__bf16)f3.x;a[13]=(__bf16)f3.y;a[14]=(__bf16)f3.z; a[15]=(__bf16)f3.w;
        // ---- B fragment (32x16 bf16): elems 0..7 @ k0, 8..15 @ k0+16
        const __bf16* bp = bslab + (lane & 15) * HPAD + kb + khalf * 8;
        v16bf bfrag = cat16(load8bf(bp), load8bf(bp + 16));
        acc = __builtin_amdgcn_wmma_f32_16x16x32_bf16(
            false, a, false, bfrag, (short)0, acc, false, false);
    }
    const float bias = bvec[col];
    float* op = P + (mt * 16 + 8 * khalf) * C_ + col;
#pragma unroll
    for (int r = 0; r < 8; ++r) op[r * C_] = acc[r] + bias;
}

// ---------------------------------------------------------------------------
// Kernel 2: persistent single-workgroup recurrence (1024 thr = 32 waves, 1 WGP)
//   h (bf16) lives in LDS (130KB); WhT streamed from L2 each step.
//   Wave w owns columns [16w, 16w+16); 8 M-tiles => 8 v8f accumulators.
//   Per step: WMMA accumulate onto P-tile, tanh, BN via LDS atomics,
//   row-norm clip, store output + write bf16 h back to LDS.
// ---------------------------------------------------------------------------
__global__ __launch_bounds__(1024) void recurrent(
    const float* __restrict__ state,  // [B, C]
    const __bf16* __restrict__ WhT,   // [C(n), C(k)]
    const float* __restrict__ gamma,  // [C]
    const float* __restrict__ beta,   // [C]
    const float* __restrict__ P,      // [T*B, C]
    float* __restrict__ out)          // [T*B, C]
{
    __shared__ __bf16 hbf[B_ * HPAD];                 // 133,120 B
    __shared__ float  colsum[C_], colsq[C_];          // BN stats -> (a, b) affine
    __shared__ float  rowsq[B_], rmask[B_];

    const int tid   = threadIdx.x;
    const int lane  = tid & 31;
    const int wave  = tid >> 5;
    const int khalf = lane >> 4;
    const int nn    = wave * 16 + (lane & 15);        // this lane's column

    // init h from initial state, zero stats
    for (int i = tid; i < B_ * C_; i += 1024) {
        int m = i >> 9, k = i & (C_ - 1);
        hbf[m * HPAD + k] = (__bf16)state[i];
    }
    if (tid < C_) { colsum[tid] = 0.0f; colsq[tid] = 0.0f; }
    if (tid >= C_ && tid < C_ + B_) rowsq[tid - C_] = 0.0f;
    __syncthreads();

    for (int t = 0; t < T_; ++t) {
        const float* Pt = P + t * B_ * C_;
        float* Ot = out + t * B_ * C_;

        // prefetch next step's pre-activation tile into cache (256B/thread)
        if (t + 1 < T_) {
            const char* pn = (const char*)(Pt + B_ * C_) + tid * 256;
            __builtin_prefetch(pn, 0, 1);
            __builtin_prefetch(pn + 128, 0, 1);
        }

        // acc[mt] := P tile (x@Wx + b precomputed)
        v8f acc[8];
#pragma unroll
        for (int mt = 0; mt < 8; ++mt) {
            const float* pp = Pt + (mt * 16 + 8 * khalf) * C_ + nn;
#pragma unroll
            for (int r = 0; r < 8; ++r) acc[mt][r] = pp[r * C_];
        }

        // acc += h @ Wh   (A from LDS, B from L2-resident WhT)
        for (int kb = 0; kb < C_; kb += 32) {
            const __bf16* bp = WhT + nn * C_ + kb + khalf * 8;
            v16bf bfrag = cat16(load8bf(bp), load8bf(bp + 16));
#pragma unroll
            for (int mt = 0; mt < 8; ++mt) {
                const __bf16* ap =
                    hbf + (mt * 16 + (lane & 15)) * HPAD + kb + khalf * 8;
                v16bf afrag = cat16(load8bf(ap), load8bf(ap + 16));
                acc[mt] = __builtin_amdgcn_wmma_f32_16x16x32_bf16(
                    false, afrag, false, bfrag, (short)0, acc[mt], false, false);
            }
        }

        // tanh + per-column (batch) stats
#pragma unroll
        for (int mt = 0; mt < 8; ++mt) {
            float cs = 0.0f, cq = 0.0f;
#pragma unroll
            for (int r = 0; r < 8; ++r) {
                float s = fast_tanh(acc[mt][r]);
                acc[mt][r] = s;
                cs += s; cq += s * s;
            }
            atomicAdd(&colsum[nn], cs);
            atomicAdd(&colsq[nn], cq);
        }
        __syncthreads();

        // BN affine per column: a = rstd*gamma, b = beta - mu*a
        if (tid < C_) {
            float mu   = colsum[tid] * (1.0f / B_);
            float var  = colsq[tid] * (1.0f / B_) - mu * mu;
            float rstd = rsqrtf(var + 1e-5f);
            float a    = rstd * gamma[tid];
            colsum[tid] = a;
            colsq[tid]  = beta[tid] - mu * a;
        } else if (tid < C_ + B_) {
            rowsq[tid - C_] = 0.0f;
        }
        __syncthreads();

        // normalize + per-row sum of squares
        {
            float ca = colsum[nn], cb = colsq[nn];
#pragma unroll
            for (int mt = 0; mt < 8; ++mt) {
#pragma unroll
                for (int r = 0; r < 8; ++r) {
                    float sn = acc[mt][r] * ca + cb;
                    acc[mt][r] = sn;
                    atomicAdd(&rowsq[mt * 16 + 8 * khalf + r], sn * sn);
                }
            }
        }
        __syncthreads();

        // row clip mask; clear column stats for next step
        if (tid < B_) {
            float nr = sqrtf(rowsq[tid]);
            rmask[tid] = (nr > 25.0f) ? (25.0f / nr) : 1.0f;
        }
        if (tid < C_) { colsum[tid] = 0.0f; colsq[tid] = 0.0f; }
        __syncthreads();

        // apply mask; write output (fp32) and new h (bf16) for next step
#pragma unroll
        for (int mt = 0; mt < 8; ++mt) {
            float* op  = Ot  + (mt * 16 + 8 * khalf) * C_   + nn;
            __bf16* hp = hbf + (mt * 16 + 8 * khalf) * HPAD + nn;
#pragma unroll
            for (int r = 0; r < 8; ++r) {
                float o = acc[mt][r] * rmask[mt * 16 + 8 * khalf + r];
                op[r * C_]   = o;
                hp[r * HPAD] = (__bf16)o;
            }
        }
        __syncthreads();
    }
}

// ---------------------------------------------------------------------------
// Kernel 3: duplicate first output into second tuple slot (full-grid copy).
// Nontemporal store: don't evict L2-resident working set with the copy.
// ---------------------------------------------------------------------------
__global__ void dup_out(const float4* __restrict__ src,
                        float4* __restrict__ dst, int n4) {
    int i = blockIdx.x * blockDim.x + threadIdx.x;
    if (i < n4) {
        float4 v = src[i];
        __builtin_nontemporal_store(v.x, &dst[i].x);
        __builtin_nontemporal_store(v.y, &dst[i].y);
        __builtin_nontemporal_store(v.z, &dst[i].z);
        __builtin_nontemporal_store(v.w, &dst[i].w);
    }
}

// ---------------------------------------------------------------------------
extern "C" void kernel_launch(void* const* d_in, const int* in_sizes, int n_in,
                              void* d_out, int out_size, void* d_ws, size_t ws_size,
                              hipStream_t stream) {
    const float* X     = (const float*)d_in[0];   // [T,B,C]
    const float* state = (const float*)d_in[1];   // [1,B,C]
    const float* Wx    = (const float*)d_in[2];   // [C,C]
    const float* Wh    = (const float*)d_in[3];   // [C,C]
    const float* bvec  = (const float*)d_in[4];   // [C]
    const float* gamma = (const float*)d_in[5];   // [C]
    const float* beta  = (const float*)d_in[6];   // [C]

    float*  out = (float*)d_out;                  // 2*T*B*C fp32
    __bf16* WxT = (__bf16*)d_ws;                  // 512 KB
    __bf16* WhT = (__bf16*)((char*)d_ws + (size_t)C_ * C_ * 2); // 512 KB
    float*  P   = out + TBC_;  // reuse 2nd output slot as fp32 scratch for x@Wx+b

    // 0) weights -> transposed bf16 (K-contiguous B-operand layout)
    convert_weights<<<(C_ * C_) / 256, 256, 0, stream>>>(Wx, Wh, WxT, WhT);

    // 1) parallel GEMM over all timesteps: P = X @ Wx + b
    gemm_xwx<<<(T_ * B_ / 16) * (C_ / 16) / 8, 256, 0, stream>>>(X, WxT, bvec, P);

    // 2) serial recurrence on one persistent WGP (h in LDS, WhT in L2)
    recurrent<<<1, 1024, 0, stream>>>(state, WhT, gamma, beta, P, out);

    // 3) second tuple output = first output (overwrites P scratch)
    int n4 = TBC_ / 4;
    dup_out<<<(n4 + 255) / 256, 256, 0, stream>>>(
        (const float4*)out, (float4*)(out + TBC_), n4);
}
